// BatchNeuralMemoryV2_19516331393468
// MI455X (gfx1250) — matrix operand order repaired and checked by reading, not compiled, not run on hardware
//
#include <hip/hip_runtime.h>
#include <cmath>
#include <cstdint>

// MI455X / gfx1250: wave32, WMMA 16x16x32 bf16 -> f32.
// GEMM inner loop: fragment-native LDS layout (packed K-pairs) so every WMMA
// operand is 2x ds_load_b128; double-buffered tiles with ASYNC global->LDS
// staging (global_load_async_to_lds_b128, ASYNCcnt) so WMMAs overlap the
// next tile's fill; 1 barrier per K-step.
// Recurrent state (~80MB) is L2-resident (192MB L2); bf16 operands halve traffic.

typedef __attribute__((ext_vector_type(8)))  float        v8f;
typedef __attribute__((ext_vector_type(16))) __bf16       v16bf;
typedef __attribute__((ext_vector_type(8)))  unsigned int v8u;

#define EPSF 1e-6f

__device__ __forceinline__ float sigmoidf_(float x) { return 1.0f / (1.0f + __expf(-x)); }
__device__ __forceinline__ float siluf_(float x)    { return x * sigmoidf_(x); }

// async copy 16B global -> LDS, tracked by ASYNCcnt (no VGPR transit)
__device__ __forceinline__ void async_b128(void* lds_dst, const void* gsrc) {
    uint32_t lds = (uint32_t)(uintptr_t)lds_dst;      // low 32 bits = LDS byte address
    uint64_t ga  = (uint64_t)(uintptr_t)gsrc;
    asm volatile("global_load_async_to_lds_b128 %0, %1, off"
                 :: "v"(lds), "v"(ga) : "memory");
}
__device__ __forceinline__ void wait_async0() {
    asm volatile("s_wait_asynccnt 0x0" ::: "memory");
}

// blockDim.x == 256 assumed
__device__ __forceinline__ float blockReduceSum(float v, float* red) {
    int tid = threadIdx.x;
    red[tid] = v; __syncthreads();
    for (int o = 128; o > 0; o >>= 1) {
        if (tid < o) red[tid] += red[tid + o];
        __syncthreads();
    }
    float r = red[0];
    __syncthreads();
    return r;
}

// ---------------------------------------------------------------------------
// WMMA GEMM: C[M,N] = A x B, f32 accumulate from bf16 operands.
// MODE 0 = NT: A row-major MxK, B stored NxK (weights, "x @ W^T")
// MODE 1 = NN: A row-major MxK, B stored KxN
// MODE 2 = TN: A stored KxM,    B stored KxN
// mapA/mapB: if nonzero, stored-row r -> (r>>6)*map + (r&63)  (chunked (B,C) rows)
// EPI: 0 = none, 1 = silu.  BF16OUT: write __bf16 instead of float.
// Block = 256 threads = 8 waves; tile 64(M) x 64(N); each wave -> 16x32 of C.
// LDS tiles hold packed bf16 K-pairs: sA[m][p] = A[m][2p] | A[m][2p+1]<<16, etc.
// ---------------------------------------------------------------------------
template <int MODE, int EPI, bool BF16OUT>
__global__ __launch_bounds__(256)
void gemm_wmma(const __bf16* __restrict__ Ag, const __bf16* __restrict__ Bg,
               void* __restrict__ Cv, int M, int N, int K, int mapA, int mapB) {
    __shared__ uint32_t sA[2][64][20];   // [m][kpair], 16 pairs + 4 pad
    __shared__ uint32_t sB[2][64][20];   // [n][kpair]
    const int tid  = threadIdx.x;
    const int lane = tid & 31;
    const int w    = tid >> 5;
    const int wm   = w & 3;      // 0..3 -> M sub-tile
    const int wn   = w >> 2;     // 0..1 -> N half
    const int m0   = blockIdx.y * 64;
    const int n0   = blockIdx.x * 64;
    const int ldA  = (MODE == 2) ? M : K;
    const int ldB  = (MODE == 0) ? K : N;

    auto stage = [&](int k0, int buf) {
        // ---- A tile ----
        if (MODE != 2) {  // row-major M x K: async 16B global -> LDS
            int r   = tid >> 2;             // 0..63 (M-local)
            int c0p = (tid & 3) * 4;        // pair offset (4 pairs = 8 bf16)
            long row = (long)(m0 + r);
            if (mapA) row = (row >> 6) * (long)mapA + (row & 63);
            const __bf16* src = Ag + row * (long)ldA + k0 + c0p * 2;
            if (k0 + 32 < K) __builtin_prefetch(src + 32, 0, 3);
            async_b128(&sA[buf][r][c0p], src);
        } else {          // A stored K x M: contiguous global read, scatter 16-bit halves
            int kk = k0 + (tid >> 3);       // one K row per thread (32 rows)
            int m8 = (tid & 7) * 8;
            long row = (long)kk;
            if (mapA) row = (row >> 6) * (long)mapA + (row & 63);
            uint4 q = *reinterpret_cast<const uint4*>(Ag + row * (long)ldA + m0 + m8);
            unsigned short* dst = (unsigned short*)&sA[buf][0][0];
            int kl = kk - k0;
            uint32_t wv[4] = {q.x, q.y, q.z, q.w};
#pragma unroll
            for (int j = 0; j < 8; ++j) {
                unsigned short h = (unsigned short)(wv[j >> 1] >> ((j & 1) * 16));
                dst[(m8 + j) * 40 + kl] = h;   // row stride 20 uints = 40 halves
            }
        }
        // ---- B tile ----
        if (MODE == 0) {  // B stored N x K: async 16B global -> LDS
            int nn  = tid >> 2;             // 0..63
            int c0p = (tid & 3) * 4;
            long row = (long)(n0 + nn);
            if (mapB) row = (row >> 6) * (long)mapB + (row & 63);
            const __bf16* src = Bg + row * (long)ldB + k0 + c0p * 2;
            if (k0 + 32 < K) __builtin_prefetch(src + 32, 0, 3);
            async_b128(&sB[buf][nn][c0p], src);
        } else {          // B stored K x N: pack two K-rows into pairs
            int kp = tid >> 4;              // 0..15 pair-row
            int n4 = (tid & 15) * 4;        // 4 consecutive n
            long r0 = (long)(k0 + 2 * kp), r1 = r0 + 1;
            if (mapB) {
                r0 = (r0 >> 6) * (long)mapB + (r0 & 63);
                r1 = (r1 >> 6) * (long)mapB + (r1 & 63);
            }
            uint2 lo = *reinterpret_cast<const uint2*>(Bg + r0 * (long)ldB + n0 + n4);
            uint2 hi = *reinterpret_cast<const uint2*>(Bg + r1 * (long)ldB + n0 + n4);
            uint32_t lov[2] = {lo.x, lo.y}, hiv[2] = {hi.x, hi.y};
#pragma unroll
            for (int i = 0; i < 4; ++i) {
                uint32_t l = (lov[i >> 1] >> ((i & 1) * 16)) & 0xffffu;
                uint32_t h = (hiv[i >> 1] >> ((i & 1) * 16)) & 0xffffu;
                sB[buf][n4 + i][kp] = l | (h << 16);
            }
        }
    };

    v8f acc0 = {}; v8f acc1 = {};
    stage(0, 0);
    int buf = 0;
    for (int k0 = 0; k0 < K; k0 += 32) {
        wait_async0();       // this wave's async fills of 'buf' are complete
        __syncthreads();     // all waves' fills visible (dscnt for scalar paths)
        if (k0 + 32 < K) stage(k0 + 32, buf ^ 1);   // async fill of other buffer

        // ---- A fragment: 2x ds_load_b128 (ISA 16-bit A 16x32 layout) ----
        const int ph = (lane >> 4) * 4;      // pair offset for lane half
        const int m  = wm * 16 + (lane & 15);
        uint4 a0 = *reinterpret_cast<const uint4*>(&sA[buf][m][ph]);
        uint4 a1 = *reinterpret_cast<const uint4*>(&sA[buf][m][8 + ph]);
        v8u au; au[0]=a0.x; au[1]=a0.y; au[2]=a0.z; au[3]=a0.w;
                au[4]=a1.x; au[5]=a1.y; au[6]=a1.z; au[7]=a1.w;
        v16bf af = __builtin_bit_cast(v16bf, au);

        // ---- B fragments: 2x ds_load_b128 each + WMMA ----
        const int p0 = (lane >> 4) * 4;
#pragma unroll
        for (int ns = 0; ns < 2; ++ns) {
            int n = wn * 32 + ns * 16 + (lane & 15);
            uint4 b0 = *reinterpret_cast<const uint4*>(&sB[buf][n][p0]);
            uint4 b1 = *reinterpret_cast<const uint4*>(&sB[buf][n][p0 + 8]);
            v8u bu; bu[0]=b0.x; bu[1]=b0.y; bu[2]=b0.z; bu[3]=b0.w;
                    bu[4]=b1.x; bu[5]=b1.y; bu[6]=b1.z; bu[7]=b1.w;
            v16bf bfg = __builtin_bit_cast(v16bf, bu);
            if (ns == 0)
                acc0 = __builtin_amdgcn_wmma_f32_16x16x32_bf16(false, af, false, bfg,
                                                               (short)0, acc0, false, false);
            else
                acc1 = __builtin_amdgcn_wmma_f32_16x16x32_bf16(false, af, false, bfg,
                                                               (short)0, acc1, false, false);
        }
        buf ^= 1;
    }

    // ---- store (C/D layout: VGPR r -> M = r + 8*(lane/16), N = lane%16) ----
#pragma unroll
    for (int ns = 0; ns < 2; ++ns) {
        v8f acc = ns ? acc1 : acc0;
        int n = n0 + wn * 32 + ns * 16 + (lane & 15);
#pragma unroll
        for (int r = 0; r < 8; ++r) {
            int m = m0 + wm * 16 + r + 8 * (lane >> 4);
            float val = acc[r];
            if (EPI == 1) val = siluf_(val);
            long idx = (long)m * N + n;
            if (BF16OUT) ((__bf16*)Cv)[idx] = (__bf16)val;
            else         ((float*)Cv)[idx]  = val;
        }
    }
}

// ---------------------------------------------------------------------------
// Elementwise / small kernels
// ---------------------------------------------------------------------------
__global__ void cvt_bf16(const float* __restrict__ src, __bf16* __restrict__ dst, int n) {
    int i = blockIdx.x * 256 + threadIdx.x;
    if (i < n) dst[i] = (__bf16)src[i];
}

__global__ void init_mem(const float* __restrict__ src, float* __restrict__ w,
                         __bf16* __restrict__ wbf, float* __restrict__ surp, int n) {
    int i = blockIdx.x * 256 + threadIdx.x;
    if (i >= n) return;
    float v = src[i];
    w[i] = v;
    if (wbf) wbf[i] = (__bf16)v;
    surp[i] = 0.0f;
}

__global__ void silu_cvt(const float* __restrict__ src, __bf16* __restrict__ dst, int n) {
    int i = blockIdx.x * 256 + threadIdx.x;
    if (i < n) dst[i] = (__bf16)siluf_(src[i]);
}

__global__ void dpre_kernel(const float* __restrict__ dh, const float* __restrict__ pre0,
                            __bf16* __restrict__ dpre_bf, int n) {
    int i = blockIdx.x * 256 + threadIdx.x;
    if (i >= n) return;
    float z  = pre0[i];
    float sg = sigmoidf_(z);
    float d  = sg * (1.0f + z * (1.0f - sg));   // silu'(z)
    dpre_bf[i] = (__bf16)(dh[i] * d);
}

// per-(b,nc) dots: xf . {alpha_w, theta_w, eta_w}, 65536 elems each
__global__ __launch_bounds__(256)
void coeff_dots(const float* __restrict__ x, const float* __restrict__ aw,
                const float* __restrict__ tw, const float* __restrict__ ew,
                float* __restrict__ raw) {
    __shared__ float red[256];
    int blk = blockIdx.x;               // b*32 + nc
    int b = blk >> 5, nc = blk & 31;
    const float* xp = x + (long)b * 2048 * 1024 + (long)nc * 64 * 1024;
    float sa = 0.f, st = 0.f, se = 0.f;
    for (int i = threadIdx.x; i < 65536; i += 256) {
        float xv = xp[i];
        sa += xv * aw[i]; st += xv * tw[i]; se += xv * ew[i];
    }
    float ra = blockReduceSum(sa, red);
    float rt = blockReduceSum(st, red);
    float re = blockReduceSum(se, red);
    if (threadIdx.x == 0) {
        raw[blk * 3 + 0] = ra;
        raw[blk * 3 + 1] = rt;
        raw[blk * 3 + 2] = re;
    }
}

__global__ void coeff_finalize(const float* __restrict__ raw, float* alpha, float* beta,
                               float* eta, float* theta) {
    int nc = threadIdx.x;
    if (nc >= 32) return;
    float sa = 0.f, se = 0.f;
    for (int b = 0; b < 8; ++b) {
        sa += raw[(b * 32 + nc) * 3 + 0];
        se += raw[(b * 32 + nc) * 3 + 2];
        theta[b * 32 + nc] = sigmoidf_(raw[(b * 32 + nc) * 3 + 1]) * 0.01f;
    }
    float a = sigmoidf_(sa / 8.0f);
    alpha[nc] = a; beta[nc] = 1.0f - a;
    eta[nc] = sigmoidf_(se / 8.0f);
}

// in-place rmsnorm over rows of width 1024; optional bf16 copy
__global__ __launch_bounds__(256)
void rmsnorm_rows(float* __restrict__ t, __bf16* __restrict__ out_bf,
                  const float* __restrict__ w) {
    __shared__ float red[256];
    long base = (long)blockIdx.x * 1024;
    int tid = threadIdx.x;
    float v[4]; float ss = 0.f;
#pragma unroll
    for (int k = 0; k < 4; ++k) { v[k] = t[base + tid + 256 * k]; ss += v[k] * v[k]; }
    float tot = blockReduceSum(ss, red);
    float s = rsqrtf(tot / 1024.0f + EPSF);
#pragma unroll
    for (int k = 0; k < 4; ++k) {
        int j = tid + 256 * k;
        float o = v[k] * s * w[j];
        t[base + j] = o;
        if (out_bf) out_bf[base + j] = (__bf16)o;
    }
}

// fused: rmsnorm stats of y, pred, dpred, rmsnorm-backward dy (bf16)
__global__ __launch_bounds__(256)
void chunk_row(const float* __restrict__ y, const float* __restrict__ kf,
               const float* __restrict__ vf, const float* __restrict__ ln,
               const float* __restrict__ theta, int chunk,
               float* __restrict__ dpred, __bf16* __restrict__ dy_bf,
               float* __restrict__ srow) {
    __shared__ float red[256];
    int i = blockIdx.x;                 // 0..511 row of chunk  (i = b*64 + c)
    int b = i >> 6, c = i & 63;
    long g = ((long)b * 2048 + (long)chunk * 64 + c) * 1024;
    long yb = (long)i * 1024;
    int tid = threadIdx.x;

    float yv[4], gv[4];
    float ss = 0.f;
#pragma unroll
    for (int k = 0; k < 4; ++k) { yv[k] = y[yb + tid + 256 * k]; ss += yv[k] * yv[k]; }
    float tot = blockReduceSum(ss, red);
    float s = rsqrtf(tot / 1024.0f + EPSF);
    if (tid == 0) srow[i] = s;

    float th = theta[b * 32 + chunk];
    const float invN = 1.0f / 524288.0f;   // mean over B*C*H
    float dot = 0.f;
#pragma unroll
    for (int k = 0; k < 4; ++k) {
        int j = tid + 256 * k;
        float lnj  = ln[j];
        float pred = kf[g + j] + yv[k] * s * lnj;
        float dp   = 2.0f * (pred - vf[g + j]) * th * invN;
        dpred[yb + j] = dp;
        gv[k] = dp * lnj;
        dot += gv[k] * yv[k];
    }
    float dsum = blockReduceSum(dot, red);
    float coef = s * s * dsum * (1.0f / 1024.0f);
#pragma unroll
    for (int k = 0; k < 4; ++k) {
        int j = tid + 256 * k;
        float dy = s * (gv[k] - coef * yv[k]);
        dy_bf[yb + j] = (__bf16)dy;
    }
}

__global__ void dln_kernel(const float* __restrict__ dpred, const float* __restrict__ y,
                           const float* __restrict__ srow, float* __restrict__ dln) {
    int j = blockIdx.x * 256 + threadIdx.x;   // < 1024
    float acc = 0.f;
    for (int i = 0; i < 512; ++i)
        acc += dpred[(long)i * 1024 + j] * y[(long)i * 1024 + j] * srow[i];
    dln[j] = acc;
}

__global__ __launch_bounds__(256)
void sumsq_kernel(const float* __restrict__ src, long n, float* __restrict__ partial) {
    __shared__ float red[256];
    float s = 0.f;
    for (long i = (long)blockIdx.x * 256 + threadIdx.x; i < n; i += (long)gridDim.x * 256) {
        float v = src[i]; s += v * v;
    }
    float tot = blockReduceSum(s, red);
    if (threadIdx.x == 0) partial[blockIdx.x] = tot;
}

__global__ void clip_kernel(const float* __restrict__ partial, float* __restrict__ clip) {
    float s = 0.f;
    for (int i = 0; i < 384; ++i) s += partial[i];
    float total = sqrtf(s);
    clip[0] = fminf(1.0f / (total + 1e-6f), 1.0f);
}

__global__ void update_kernel(float* __restrict__ w, float* __restrict__ surp,
                              const float* __restrict__ g, __bf16* __restrict__ wbf,
                              const float* __restrict__ eta, const float* __restrict__ beta,
                              const float* __restrict__ clip, int chunk, int n) {
    int i = blockIdx.x * 256 + threadIdx.x;
    if (i >= n) return;
    float e = eta[chunk], bt = beta[chunk], cl = clip[0];
    float sn = e * surp[i] - g[i] * cl;
    surp[i] = sn;
    float wn = bt * w[i] + sn;
    w[i] = wn;
    if (wbf) wbf[i] = (__bf16)wn;
}

// out = q + rmsnorm(y, ln), y resides in d_out (in-place)
__global__ __launch_bounds__(256)
void final_rows(float* __restrict__ y, const float* __restrict__ q,
                const float* __restrict__ ln) {
    __shared__ float red[256];
    long base = (long)blockIdx.x * 1024;
    int tid = threadIdx.x;
    float v[4]; float ss = 0.f;
#pragma unroll
    for (int k = 0; k < 4; ++k) { v[k] = y[base + tid + 256 * k]; ss += v[k] * v[k]; }
    float tot = blockReduceSum(ss, red);
    float s = rsqrtf(tot / 1024.0f + EPSF);
#pragma unroll
    for (int k = 0; k < 4; ++k) {
        int j = tid + 256 * k;
        y[base + j] = q[base + j] + v[k] * s * ln[j];
    }
}

// ---------------------------------------------------------------------------
extern "C" void kernel_launch(void* const* d_in, const int* in_sizes, int n_in,
                              void* d_out, int out_size, void* d_ws, size_t ws_size,
                              hipStream_t stream) {
    (void)in_sizes; (void)n_in; (void)out_size; (void)ws_size;
    const float* x      = (const float*)d_in[0];
    const float* wq     = (const float*)d_in[1];
    const float* wk     = (const float*)d_in[2];
    const float* wv     = (const float*)d_in[3];
    const float* qn     = (const float*)d_in[4];
    const float* kn     = (const float*)d_in[5];
    const float* aw     = (const float*)d_in[6];
    const float* tw     = (const float*)d_in[7];
    const float* ew     = (const float*)d_in[8];
    const float* mw0_in = (const float*)d_in[9];
    const float* mw1_in = (const float*)d_in[10];
    const float* mln_in = (const float*)d_in[11];
    float* out = (float*)d_out;

    char* ws = (char*)d_ws;
    size_t off = 0;
    auto alloc = [&](size_t bytes) -> void* {
        void* p = ws + off;
        off = (off + bytes + 255) & ~(size_t)255;
        return p;
    };

    __bf16* x_bf    = (__bf16*)alloc(16777216ull * 2);
    __bf16* wq_bf   = (__bf16*)alloc(1048576ull * 2);
    __bf16* wk_bf   = (__bf16*)alloc(1048576ull * 2);
    __bf16* wv_bf   = (__bf16*)alloc(1048576ull * 2);
    float*  k_f     = (float*) alloc(16777216ull * 4);
    __bf16* k_bf    = (__bf16*)alloc(16777216ull * 2);
    float*  v_f     = (float*) alloc(16777216ull * 4);   // reused as hq_bf after scan
    float*  q_f     = (float*) alloc(16777216ull * 4);
    __bf16* q_bf    = (__bf16*)alloc(16777216ull * 2);
    float*  mw0     = (float*) alloc(2097152ull * 4);
    float*  mw1     = (float*) alloc(2097152ull * 4);
    float*  mln     = (float*) alloc(1024 * 4);
    float*  s0      = (float*) alloc(2097152ull * 4);
    float*  s1      = (float*) alloc(2097152ull * 4);
    float*  sln     = (float*) alloc(1024 * 4);
    __bf16* w0_bf   = (__bf16*)alloc(2097152ull * 2);
    __bf16* w1_bf   = (__bf16*)alloc(2097152ull * 2);
    float*  pre0    = (float*) alloc(1048576ull * 4);
    __bf16* h_bf    = (__bf16*)alloc(1048576ull * 2);
    float*  ybuf    = (float*) alloc(524288ull * 4);
    float*  srow    = (float*) alloc(512 * 4);
    float*  dpred   = (float*) alloc(524288ull * 4);
    __bf16* dy_bf   = (__bf16*)alloc(524288ull * 2);
    float*  dh      = (float*) alloc(1048576ull * 4);
    __bf16* dpre_bf = (__bf16*)alloc(1048576ull * 2);
    float*  dln     = (float*) alloc(1024 * 4);
    float*  gW0     = (float*) alloc(2097152ull * 4);
    float*  gW1     = (float*) alloc(2097152ull * 4);
    float*  raw     = (float*) alloc(256 * 3 * 4);
    float*  alphab  = (float*) alloc(32 * 4);
    float*  betab   = (float*) alloc(32 * 4);
    float*  etab    = (float*) alloc(32 * 4);
    float*  thetab  = (float*) alloc(256 * 4);
    float*  partial = (float*) alloc(384 * 4);
    float*  clipb   = (float*) alloc(4);
    __bf16* hq_bf   = (__bf16*)v_f;   // 16384x2048 bf16 == 64MB, exact alias of dead v_f
    (void)alphab;

    // ---- setup: bf16 shadows, memory init, coefficients ----
    cvt_bf16<<<65536, 256, 0, stream>>>(x, x_bf, 16777216);
    cvt_bf16<<<4096, 256, 0, stream>>>(wq, wq_bf, 1048576);
    cvt_bf16<<<4096, 256, 0, stream>>>(wk, wk_bf, 1048576);
    cvt_bf16<<<4096, 256, 0, stream>>>(wv, wv_bf, 1048576);
    init_mem<<<8192, 256, 0, stream>>>(mw0_in, mw0, w0_bf, s0, 2097152);
    init_mem<<<8192, 256, 0, stream>>>(mw1_in, mw1, w1_bf, s1, 2097152);
    init_mem<<<4, 256, 0, stream>>>(mln_in, mln, (__bf16*)nullptr, sln, 1024);
    coeff_dots<<<256, 256, 0, stream>>>(x, aw, tw, ew, raw);
    coeff_finalize<<<1, 32, 0, stream>>>(raw, alphab, betab, etab, thetab);

    // ---- projections: k, v, q (silu epilogue, then rmsnorm rows for k,q) ----
    gemm_wmma<0, 1, false><<<dim3(16, 256), 256, 0, stream>>>(x_bf, wk_bf, k_f, 16384, 1024, 1024, 0, 0);
    rmsnorm_rows<<<16384, 256, 0, stream>>>(k_f, k_bf, kn);
    gemm_wmma<0, 1, false><<<dim3(16, 256), 256, 0, stream>>>(x_bf, wv_bf, v_f, 16384, 1024, 1024, 0, 0);
    gemm_wmma<0, 1, false><<<dim3(16, 256), 256, 0, stream>>>(x_bf, wq_bf, q_f, 16384, 1024, 1024, 0, 0);
    rmsnorm_rows<<<16384, 256, 0, stream>>>(q_f, q_bf, qn);

    // ---- sequential scan over 32 chunks ----
    for (int ch = 0; ch < 32; ++ch) {
        const __bf16* kc = k_bf + (long)ch * 64 * 1024;   // chunk base; row map (r>>6)*2048+(r&63)
        // forward: pre0 = Kc @ W0^T ; h = silu(pre0) ; y = h @ W1^T
        gemm_wmma<0, 0, false><<<dim3(32, 8), 256, 0, stream>>>(kc, w0_bf, pre0, 512, 2048, 1024, 2048, 0);
        silu_cvt<<<4096, 256, 0, stream>>>(pre0, h_bf, 1048576);
        gemm_wmma<0, 0, false><<<dim3(16, 8), 256, 0, stream>>>(h_bf, w1_bf, ybuf, 512, 1024, 2048, 0, 0);
        // loss + rmsnorm backward -> dpred, dy
        chunk_row<<<512, 256, 0, stream>>>(ybuf, k_f, v_f, mln, thetab, ch, dpred, dy_bf, srow);
        dln_kernel<<<4, 256, 0, stream>>>(dpred, ybuf, srow, dln);
        // dW1 = dy^T @ h ; dh = dy @ W1 ; dpre = dh * silu'(pre0) ; dW0 = dpre^T @ Kc
        gemm_wmma<2, 0, false><<<dim3(32, 16), 256, 0, stream>>>(dy_bf, h_bf, gW1, 1024, 2048, 512, 0, 0);
        gemm_wmma<1, 0, false><<<dim3(32, 8), 256, 0, stream>>>(dy_bf, w1_bf, dh, 512, 2048, 1024, 0, 0);
        dpre_kernel<<<4096, 256, 0, stream>>>(dh, pre0, dpre_bf, 1048576);
        gemm_wmma<2, 0, false><<<dim3(16, 32), 256, 0, stream>>>(dpre_bf, kc, gW0, 2048, 1024, 512, 0, 2048);
        // global grad-norm clip (deterministic two-stage reduction)
        sumsq_kernel<<<128, 256, 0, stream>>>(gW1, 2097152L, partial);
        sumsq_kernel<<<128, 256, 0, stream>>>(gW0, 2097152L, partial + 128);
        sumsq_kernel<<<128, 256, 0, stream>>>(dln, 1024L, partial + 256);
        clip_kernel<<<1, 1, 0, stream>>>(partial, clipb);
        // surp = eta*surp - g*clip ; mem = beta*mem + surp  (refresh bf16 shadows)
        update_kernel<<<8192, 256, 0, stream>>>(mw0, s0, gW0, w0_bf, etab, betab, clipb, ch, 2097152);
        update_kernel<<<8192, 256, 0, stream>>>(mw1, s1, gW1, w1_bf, etab, betab, clipb, ch, 2097152);
        update_kernel<<<4, 256, 0, stream>>>(mln, sln, dln, (__bf16*)nullptr, etab, betab, clipb, ch, 1024);
    }

    // ---- retrieval with final memory: out = q + rmsnorm(silu(q@W0^T)@W1^T, ln) ----
    gemm_wmma<0, 1, true><<<dim3(32, 256), 256, 0, stream>>>(q_bf, w0_bf, hq_bf, 16384, 2048, 1024, 0, 0);
    gemm_wmma<0, 0, false><<<dim3(16, 256), 256, 0, stream>>>(hq_bf, w1_bf, out, 16384, 1024, 2048, 0, 0);
    final_rows<<<16384, 256, 0, stream>>>(out, q_f, mln);
}
